// SSWL_60275571032123
// MI455X (gfx1250) — matrix operand order repaired and checked by reading, not compiled, not run on hardware
//
#include <hip/hip_runtime.h>

// ---------------------------------------------------------------------------
// MI455X (gfx1250, wave32) implementation.
// bf16 WMMA operands (v_wmma_f32_16x16x32_bf16), f32 accumulate, f32 state.
// CDNA5 data movement: TDM (tensor_load_to_lds) stages the weight matrices
// with hardware row-padding; global_load_async_to_lds_b128 stages the per-
// block activation tile. Problem is bandwidth-bound (~1.4 GB total traffic
// @23.3 TB/s vs ~43 GFLOP), so bf16 operands + fused epilogues dominate.
// ---------------------------------------------------------------------------

typedef __bf16 bf16;
typedef __attribute__((ext_vector_type(16))) __bf16 v16bf;
typedef __attribute__((ext_vector_type(8)))  float  v8f;
typedef __attribute__((ext_vector_type(4)))  float  f32x4;
typedef __attribute__((ext_vector_type(2)))  float  f32x2;
typedef __attribute__((ext_vector_type(4)))  unsigned int u32x4;
typedef __attribute__((ext_vector_type(8)))  unsigned int u32x8;

#define Bq 64
#define Nq 64
#define Dq 128

union FragBF {
    v16bf v;
    bf16  e[16];
    f32x4 q[2];
};

union Pack2BF {
    unsigned int u;
    bf16 e[2];
};

static __device__ inline v8f wmma_bf16(v16bf a, v16bf b, v8f c) {
    return __builtin_amdgcn_wmma_f32_16x16x32_bf16(
        /*neg_a=*/false, a, /*neg_b=*/false, b,
        /*c_mod=*/(short)0, c, /*reuse_a=*/false, /*reuse_b=*/false);
}

// A-matrix fragment (16x32 bf16, MxK). lane<16: M=lane, K chunks {0..7,16..23};
// lane>=16: same M, K chunks {8..15,24..31}. Row must be k-contiguous, 16B aligned.
static __device__ inline v16bf load_frag_a(const bf16* row, int ks, int lane) {
    FragBF f;
    int k0 = (lane & 16) ? 8 : 0;
    f.q[0] = *(const f32x4*)(row + ks + k0);
    f.q[1] = *(const f32x4*)(row + ks + k0 + 16);
    return f.v;
}

// B-matrix fragment (32x16 bf16, KxN): column n=lane&15; lane<16: K chunks
// {0..7,16..23}; lane>=16: K chunks {8..15,24..31}. col ptr must be k-contiguous.
static __device__ inline v16bf load_frag_b(const bf16* col, int ks, int lane) {
    FragBF f;
    int kb = ks + ((lane & 16) ? 8 : 0);
    f.q[0] = *(const f32x4*)(col + kb);
    f.q[1] = *(const f32x4*)(col + kb + 16);
    return f.v;
}

// ---------------------------------------------------------------------------
// TDM: DMA a 128x128 bf16 matrix (row-major, contiguous) from global into LDS,
// inserting 4 DWORDs (16B) of padding after every 64 DWORDs (= one 128-elem
// row) -> LDS rows of 136 bf16, bank-conflict-free for ds_load_b128 fragments.
// D# per CDNA5 ISA ch.8: group0 {count, lds_addr, global_addr, type=2},
// group1 {data_size=1(2B), pad_enable, pad_interval=5(64dw), pad_amount=3(4dw),
//         tensor_dim0/1=128, tile_dim0/1=128, dim0_stride=128}.
// ---------------------------------------------------------------------------
static __device__ __forceinline__ void tdm_load_128x128_pad(const bf16* gsrc,
                                                            bf16* ldst) {
    unsigned long long ga = (unsigned long long)(size_t)gsrc;
    u32x4 g0;
    g0[0] = 1u;                                  // count=1, user descriptor
    g0[1] = (unsigned)(size_t)ldst;              // lds_addr (LDS byte offset)
    g0[2] = (unsigned)ga;                        // global_addr[31:0]
    g0[3] = (unsigned)(ga >> 32) | (2u << 30);   // global_addr[56:32] | type=2
    u32x8 g1;
    g1[0] = (1u << 16) | (1u << 20) | (5u << 22) | (3u << 25);
    g1[1] = 128u << 16;    // tensor_dim0[15:0] at bits63:48
    g1[2] = 128u << 16;    // tensor_dim1[15:0] at bits95:80
    g1[3] = 128u << 16;    // tile_dim0 at bits127:112
    g1[4] = 128u;          // tile_dim1 at bits143:128
    g1[5] = 128u;          // tensor_dim0_stride[31:0] at bits191:160
    g1[6] = 0u;
    g1[7] = 0u;
    asm volatile("tensor_load_to_lds %0, %1" :: "s"(g0), "s"(g1) : "memory");
}

// Async copy 16B from global to LDS (per-lane addresses), tracked by ASYNCcnt.
static __device__ __forceinline__ void async_copy_b128(const void* gsrc,
                                                       void* ldst) {
    unsigned lds = (unsigned)(size_t)ldst;   // low 32 bits = LDS byte offset
    asm volatile("global_load_async_to_lds_b128 %0, %1, off"
                 :: "v"(lds), "v"(gsrc) : "memory");
}

static __device__ __forceinline__ void wait_asynccnt0() {
    asm volatile("s_wait_asynccnt 0" ::: "memory");
}

// ---------------------------------------------------------------------------
// K0: weight transpose f32[k][n] -> bf16[n][k] (so B-fragments are contiguous)
// ---------------------------------------------------------------------------
__global__ void transpose_bf16_kernel(const float* __restrict__ W,
                                      bf16* __restrict__ Wt) {
    int id = blockIdx.x * blockDim.x + threadIdx.x;
    if (id >= Dq * Dq) return;
    int n = id / Dq, k = id % Dq;
    Wt[n * Dq + k] = (bf16)W[k * Dq + n];
}

// ---------------------------------------------------------------------------
// K1: xe = x_emb[x] * nodemask  (f32 + bf16 copies)
// ---------------------------------------------------------------------------
__global__ void encode_kernel(const float* __restrict__ x_emb,
                              const int* __restrict__ x,
                              const int* __restrict__ nodemask,
                              float* __restrict__ xe,
                              bf16* __restrict__ xebf) {
    int id = blockIdx.x * blockDim.x + threadIdx.x;  // B*N*D
    int d = id & (Dq - 1);
    int bn = id >> 7;
    float v = nodemask[bn] ? x_emb[x[bn] * Dq + d] : 0.f;
    xe[id] = v;
    xebf[id] = (bf16)v;
}

// ---------------------------------------------------------------------------
// K2: single GEMM + bias, f32 out.  out[r][n] = sum_k A[r][k]*Bt[n][k] + b[n]
// ---------------------------------------------------------------------------
__global__ __launch_bounds__(256)
void gemm_bias_kernel(const bf16* __restrict__ A, const bf16* __restrict__ Bt,
                      const float* __restrict__ bias, float* __restrict__ out) {
    int rowBase = blockIdx.x * 64;
    int tid = threadIdx.x, w = tid >> 5, lane = tid & 31;
    int l15 = lane & 15, hi = lane >> 4;
    for (int q = 0; q < 4; ++q) {
        int tile = w * 4 + q;          // 0..31
        int ti = tile >> 3, tj = tile & 7;
        v8f c = {};
        const bf16* arow = A + (size_t)(rowBase + ti * 16 + l15) * Dq;
        const bf16* bcol = Bt + (size_t)(tj * 16 + l15) * Dq;
        for (int ks = 0; ks < Dq; ks += 32) {
            v16bf af = load_frag_a(arow, ks, lane);
            v16bf bf = load_frag_b(bcol, ks, lane);
            c = wmma_bf16(af, bf, c);
        }
        int n = tj * 16 + l15;
        float bv = bias[n];
        for (int r = 0; r < 8; ++r) {
            int m = ti * 16 + r + hi * 8;
            out[(size_t)(rowBase + m) * Dq + n] = c[r] + bv;
        }
    }
}

// ---------------------------------------------------------------------------
// K3: X0[b,i,j,d] = xe[b,j,d]*xl[b,i,d]*tf_emb[tf[b,i,j],d]*tm   (f32)
//     A [b,i,j,d] = ea_emb[ea[b,i,j],d]*am                       (bf16)
// ---------------------------------------------------------------------------
__global__ void build_XA_kernel(const float* __restrict__ tf_emb,
                                const float* __restrict__ ea_emb,
                                const int* __restrict__ tuplefeat,
                                const int* __restrict__ ea,
                                const int* __restrict__ adjmask,
                                const int* __restrict__ nodemask,
                                const float* __restrict__ xe,
                                const float* __restrict__ xl,
                                float* __restrict__ X,
                                bf16* __restrict__ Abf) {
    size_t id = (size_t)blockIdx.x * blockDim.x + threadIdx.x;  // B*N*N*(D/4)
    int dq = (int)(id & 31);
    size_t cell = id >> 5;
    int d = dq << 2;
    int b = (int)(cell >> 12);
    int ij = (int)(cell & 4095);
    int i = ij >> 6, j = ij & 63;
    bool tm = nodemask[b * Nq + i] && nodemask[b * Nq + j];
    bool am = tm && adjmask[cell];
    f32x4 tf4 = *(const f32x4*)(tf_emb + (size_t)tuplefeat[cell] * Dq + d);
    f32x4 xe4 = *(const f32x4*)(xe + (size_t)(b * Nq + j) * Dq + d);
    f32x4 xl4 = *(const f32x4*)(xl + (size_t)(b * Nq + i) * Dq + d);
    f32x4 zero = {0.f, 0.f, 0.f, 0.f};
    f32x4 xv = tm ? (xe4 * xl4 * tf4) : zero;
    *(f32x4*)(X + cell * Dq + d) = xv;
    f32x4 ea4 = *(const f32x4*)(ea_emb + (size_t)ea[cell] * Dq + d);
    union { bf16 e[4]; f32x2 f2; } u;
    for (int c = 0; c < 4; ++c) u.e[c] = am ? (bf16)ea4[c] : (bf16)0.f;
    *(f32x2*)(Abf + cell * Dq + d) = u.f2;
}

// ---------------------------------------------------------------------------
// K4/K6: per-d batched 64x64x64 GEMM:
//   out[b,i,j,d] = sum_k L[b,i,k,d] * R[b,k,j,d]     (out bf16)
// One block per (b, 2-wide d slice). Staging needs f32->bf16 conversion and a
// transpose, so it goes through VALU + ds_store (DMA cannot convert).
// ---------------------------------------------------------------------------
__global__ __launch_bounds__(256)
void einsum_kernel(const void* __restrict__ Lp, int Lf32,
                   const void* __restrict__ Rp, int Rf32,
                   bf16* __restrict__ out) {
    __shared__ __align__(16) bf16 Ls[2][64][72];  // [d][i][k], padded rows
    __shared__ __align__(16) bf16 Rs[2][64][72];  // [d][j][k]
    int bx = blockIdx.x;            // B * (D/2)
    int b = bx >> 6;
    int d0 = (bx & 63) << 1;
    int tid = threadIdx.x;

    for (int idx = tid; idx < 4096; idx += 256) {
        int i = idx >> 6, k = idx & 63;
        size_t g = (((size_t)(b * Nq + i)) * Nq + k) * Dq + d0;
        bf16 v0, v1;
        if (Lf32) {
            f32x2 v = *(const f32x2*)((const float*)Lp + g);
            v0 = (bf16)v[0]; v1 = (bf16)v[1];
        } else {
            Pack2BF p; p.u = *(const unsigned*)((const bf16*)Lp + g);
            v0 = p.e[0]; v1 = p.e[1];
        }
        Ls[0][i][k] = v0; Ls[1][i][k] = v1;
    }
    for (int idx = tid; idx < 4096; idx += 256) {
        int k = idx >> 6, j = idx & 63;
        size_t g = (((size_t)(b * Nq + k)) * Nq + j) * Dq + d0;
        bf16 v0, v1;
        if (Rf32) {
            f32x2 v = *(const f32x2*)((const float*)Rp + g);
            v0 = (bf16)v[0]; v1 = (bf16)v[1];
        } else {
            Pack2BF p; p.u = *(const unsigned*)((const bf16*)Rp + g);
            v0 = p.e[0]; v1 = p.e[1];
        }
        Rs[0][j][k] = v0; Rs[1][j][k] = v1;
    }
    __syncthreads();

    int w = tid >> 5, lane = tid & 31, l15 = lane & 15, hi = lane >> 4;
    for (int q = 0; q < 4; ++q) {
        int tile = w * 4 + q;                  // 0..31 = (d, 4x4 tiles)
        int dd = tile >> 4;
        int rem = tile & 15;
        int ti = rem >> 2, tj = rem & 3;
        v8f c = {};
        const bf16* arow = &Ls[dd][ti * 16 + l15][0];
        const bf16* bcol = &Rs[dd][tj * 16 + l15][0];
        for (int ks = 0; ks < 64; ks += 32) {
            v16bf af = load_frag_a(arow, ks, lane);
            v16bf bf = load_frag_b(bcol, ks, lane);
            c = wmma_bf16(af, bf, c);
        }
        for (int r = 0; r < 8; ++r) {
            int m = ti * 16 + r + hi * 8;
            int n = tj * 16 + l15;
            size_t g = (((size_t)(b * Nq + m)) * Nq + n) * Dq + d0 + dd;
            out[g] = (bf16)c[r];
        }
    }
}

// ---------------------------------------------------------------------------
// K5/K7: fused 2-layer MLP with residual + tuple mask:
//   H = relu(M @ W1 + b1); X = (X + H @ W2 + b2) * tm
// 64 rows x 128 cols per block. W1/W2 staged via TDM with hardware padding;
// the m tile staged via async-to-LDS; H staged in LDS bf16 between stages.
// ---------------------------------------------------------------------------
__global__ __launch_bounds__(256)
void mlp2_resid_kernel(const bf16* __restrict__ Min,
                       const bf16* __restrict__ W1t, const float* __restrict__ b1,
                       const bf16* __restrict__ W2t, const float* __restrict__ b2,
                       float* __restrict__ X,
                       const int* __restrict__ nodemask) {
    __shared__ __align__(16) bf16 Ms[64][136];
    __shared__ __align__(16) bf16 Hs[64][136];
    __shared__ __align__(16) bf16 W1s[128][136];
    __shared__ __align__(16) bf16 W2s[128][136];
    int rowBase = blockIdx.x * 64;
    int tid = threadIdx.x, w = tid >> 5, lane = tid & 31;
    int l15 = lane & 15, hi = lane >> 4;

    // --- staging: TDM for weights (wave 0), async-to-LDS for the m tile ---
    if (w == 0) {
        tdm_load_128x128_pad(W1t, &W1s[0][0]);
        tdm_load_128x128_pad(W2t, &W2s[0][0]);
    }
    for (int c = tid; c < 512; c += 256) {     // 64 rows x 8 chunks of 16B
        int row = c >> 3, ch = (c & 7) * 8;
        async_copy_b128(Min + (size_t)(rowBase + row) * Dq + ch, &Ms[row][ch]);
    }
    wait_asynccnt0();
    if (w == 0) __builtin_amdgcn_s_wait_tensorcnt(0);
    __syncthreads();

    // --- stage 1: H = relu(M @ W1 + b1) ---
    for (int q = 0; q < 4; ++q) {
        int tile = w * 4 + q;
        int ti = tile >> 3, tj = tile & 7;
        v8f c = {};
        const bf16* arow = &Ms[ti * 16 + l15][0];
        const bf16* bcol = &W1s[tj * 16 + l15][0];
        for (int ks = 0; ks < Dq; ks += 32) {
            v16bf af = load_frag_a(arow, ks, lane);
            v16bf bf = load_frag_b(bcol, ks, lane);
            c = wmma_bf16(af, bf, c);
        }
        int n = tj * 16 + l15;
        float bv = b1[n];
        for (int r = 0; r < 8; ++r) {
            int m = ti * 16 + r + hi * 8;
            Hs[m][n] = (bf16)fmaxf(c[r] + bv, 0.f);
        }
    }
    __syncthreads();

    // --- stage 2: X = (X + H @ W2 + b2) * tm ---
    for (int q = 0; q < 4; ++q) {
        int tile = w * 4 + q;
        int ti = tile >> 3, tj = tile & 7;
        v8f c = {};
        const bf16* arow = &Hs[ti * 16 + l15][0];
        const bf16* bcol = &W2s[tj * 16 + l15][0];
        for (int ks = 0; ks < Dq; ks += 32) {
            v16bf af = load_frag_a(arow, ks, lane);
            v16bf bf = load_frag_b(bcol, ks, lane);
            c = wmma_bf16(af, bf, c);
        }
        int n = tj * 16 + l15;
        float bv = b2[n];
        for (int r = 0; r < 8; ++r) {
            int m = ti * 16 + r + hi * 8;
            int gr = rowBase + m;                 // flattened (b,i,j)
            int bb = gr >> 12, ii = (gr >> 6) & 63, jj = gr & 63;
            bool tm = nodemask[bb * Nq + ii] && nodemask[bb * Nq + jj];
            size_t idx = (size_t)gr * Dq + n;
            float xo = X[idx];
            X[idx] = tm ? (xo + c[r] + bv) : 0.f;
        }
    }
}

// ---------------------------------------------------------------------------
// K8: masked lpool over i, masked gpool over j, head MLP (128->128->1).
// ---------------------------------------------------------------------------
__global__ __launch_bounds__(128)
void pool_head_kernel(const float* __restrict__ X,
                      const int* __restrict__ nodemask,
                      const float* __restrict__ hW1, const float* __restrict__ hb1,
                      const float* __restrict__ hW2, const float* __restrict__ hb2,
                      float* __restrict__ out) {
    __shared__ int   nms[Nq];
    __shared__ float hgs[Dq];
    __shared__ float red[Dq];
    int b = blockIdx.x, t = threadIdx.x;
    if (t < Nq) nms[t] = nodemask[b * Nq + t];
    __syncthreads();
    int ncnt = 0;
    for (int k = 0; k < Nq; ++k) ncnt += (nms[k] ? 1 : 0);
    float fnc = (float)(ncnt > 0 ? ncnt : 1);
    float hg = 0.f;
    for (int j = 0; j < Nq; ++j) {
        if (!nms[j]) continue;
        float s = 0.f;
        for (int i = 0; i < Nq; ++i)
            s += X[(((size_t)(b * Nq + i)) * Nq + j) * Dq + t];
        hg += s / fnc;   // cnt[b,j] = max(nm[j]*ncnt,1) = ncnt here
    }
    hg /= fnc;
    hgs[t] = hg;
    __syncthreads();
    float h = hb1[t];
    for (int d = 0; d < Dq; ++d) h += hgs[d] * hW1[d * Dq + t];
    h = fmaxf(h, 0.f);
    red[t] = h * hW2[t];
    __syncthreads();
    for (int s = Dq / 2; s > 0; s >>= 1) {
        if (t < s) red[t] += red[t + s];
        __syncthreads();
    }
    if (t == 0) out[b] = red[0] + hb2[0];
}

// ---------------------------------------------------------------------------
extern "C" void kernel_launch(void* const* d_in, const int* in_sizes, int n_in,
                              void* d_out, int out_size, void* d_ws, size_t ws_size,
                              hipStream_t stream) {
    (void)in_sizes; (void)n_in; (void)out_size; (void)ws_size;
    const float* x_emb  = (const float*)d_in[0];
    const float* ea_emb = (const float*)d_in[1];
    const float* tf_emb = (const float*)d_in[2];
    const float* Wt     = (const float*)d_in[3];
    const float* bt     = (const float*)d_in[4];
    const float* nW1    = (const float*)d_in[5];
    const float* nb1    = (const float*)d_in[6];
    const float* nW2    = (const float*)d_in[7];
    const float* nb2    = (const float*)d_in[8];
    const float* cW1    = (const float*)d_in[9];
    const float* cb1    = (const float*)d_in[10];
    const float* cW2    = (const float*)d_in[11];
    const float* cb2    = (const float*)d_in[12];
    const float* hW1    = (const float*)d_in[13];
    const float* hb1    = (const float*)d_in[14];
    const float* hW2    = (const float*)d_in[15];
    const float* hb2    = (const float*)d_in[16];
    const int* x         = (const int*)d_in[17];
    const int* nodemask  = (const int*)d_in[18];
    const int* ea        = (const int*)d_in[19];
    const int* adjmask   = (const int*)d_in[20];
    const int* tuplefeat = (const int*)d_in[21];
    float* out = (float*)d_out;

    // workspace layout
    char* ws = (char*)d_ws;
    size_t off = 0;
    auto alloc = [&](size_t bytes) { size_t o = off; off += (bytes + 255) & ~(size_t)255; return o; };
    const size_t nBNND = (size_t)Bq * Nq * Nq * Dq;   // 33,554,432
    const size_t nBND  = (size_t)Bq * Nq * Dq;        // 524,288
    float* X     = (float*)(ws + alloc(nBNND * 4));
    bf16*  Abf   = (bf16*) (ws + alloc(nBNND * 2));
    bf16*  Mbf   = (bf16*) (ws + alloc(nBNND * 2));
    float* xe    = (float*)(ws + alloc(nBND * 4));
    bf16*  xebf  = (bf16*) (ws + alloc(nBND * 2));
    float* xl    = (float*)(ws + alloc(nBND * 4));
    bf16*  Wt_t  = (bf16*) (ws + alloc((size_t)Dq * Dq * 2));
    bf16*  nW1t  = (bf16*) (ws + alloc((size_t)Dq * Dq * 2));
    bf16*  nW2t  = (bf16*) (ws + alloc((size_t)Dq * Dq * 2));
    bf16*  cW1t  = (bf16*) (ws + alloc((size_t)Dq * Dq * 2));
    bf16*  cW2t  = (bf16*) (ws + alloc((size_t)Dq * Dq * 2));

    // K0: weight prep (transpose+convert to bf16)
    transpose_bf16_kernel<<<64, 256, 0, stream>>>(Wt,  Wt_t);
    transpose_bf16_kernel<<<64, 256, 0, stream>>>(nW1, nW1t);
    transpose_bf16_kernel<<<64, 256, 0, stream>>>(nW2, nW2t);
    transpose_bf16_kernel<<<64, 256, 0, stream>>>(cW1, cW1t);
    transpose_bf16_kernel<<<64, 256, 0, stream>>>(cW2, cW2t);

    // K1: xe
    encode_kernel<<<(int)(nBND / 256), 256, 0, stream>>>(x_emb, x, nodemask, xe, xebf);

    // K2: xl = xe @ Wt + bt   (4096 x 128 x 128)
    gemm_bias_kernel<<<(Bq * Nq) / 64, 256, 0, stream>>>(xebf, Wt_t, bt, xl);

    // K3: build X0 (f32) and A (bf16)
    build_XA_kernel<<<(int)(nBNND / 4 / 256), 256, 0, stream>>>(
        tf_emb, ea_emb, tuplefeat, ea, adjmask, nodemask, xe, xl, X, Abf);

    // K4: nested einsum  m[b,i,j,d] = sum_k X[b,i,k,d] * A[b,k,j,d]
    einsum_kernel<<<Bq * (Dq / 2), 256, 0, stream>>>(X, 1, Abf, 0, Mbf);
    // K5: X = (X + mlp_nested(m)) * tm
    mlp2_resid_kernel<<<(int)(nBNND / Dq / 64), 256, 0, stream>>>(
        Mbf, nW1t, nb1, nW2t, nb2, X, nodemask);

    // K6: cross einsum  m[b,i,j,d] = sum_k A[b,i,k,d] * X[b,k,j,d]
    einsum_kernel<<<Bq * (Dq / 2), 256, 0, stream>>>(Abf, 0, X, 1, Mbf);
    // K7: X = (X + mlp_cross(m)) * tm
    mlp2_resid_kernel<<<(int)(nBNND / Dq / 64), 256, 0, stream>>>(
        Mbf, cW1t, cb1, cW2t, cb2, X, nodemask);

    // K8: pooling + head
    pool_head_kernel<<<Bq, 128, 0, stream>>>(X, nodemask, hW1, hb1, hW2, hb2, out);
}
